// MultiheadAttention_89996744721068
// MI455X (gfx1250) — matrix operand order repaired and checked by reading, not compiled, not run on hardware
//
#include <hip/hip_runtime.h>
#include <hip/hip_bf16.h>

// ---------------------------------------------------------------------------
// MI455X (gfx1250) fused multi-head-attention block.
//   N=32768 tokens, D=1024, H=16 heads, DK=64 (attention over the d_k axis).
// bf16 WMMA (v_wmma_f32_16x16x32_bf16, fp32 accumulate) for all four big
// GEMMs and the per-row batched attention GEMMs; softmax + LayerNorm in fp32.
// GEMM: double-buffered LDS, one barrier per K-step, L2 prefetch.
// Attention: K-dim padding lives in zero-filled LDS so every fragment half is
// a straight ds_load_b128 (no exec-masked zeroing).
// ---------------------------------------------------------------------------

typedef __attribute__((ext_vector_type(16))) __bf16        v16bf;
typedef __attribute__((ext_vector_type(8)))  float         v8f;
typedef __attribute__((ext_vector_type(8)))  unsigned int  v8u;

#define NROWS 32768
#define DMODEL 1024

__device__ __forceinline__ unsigned short f2bf(float f) {
  unsigned int u = __float_as_uint(f);
  unsigned int r = u + 0x7fffu + ((u >> 16) & 1u);   // round-to-nearest-even
  return (unsigned short)(r >> 16);
}

__device__ __forceinline__ v16bf pack_frag(uint4 lo, uint4 hi) {
  v8u u;
  u[0] = lo.x; u[1] = lo.y; u[2] = lo.z; u[3] = lo.w;
  u[4] = hi.x; u[5] = hi.y; u[6] = hi.z; u[7] = hi.w;
  return __builtin_bit_cast(v16bf, u);
}

// ---------------------------------------------------------------------------
// fp32 -> bf16 conversion, 4 elements / thread
// ---------------------------------------------------------------------------
__global__ __launch_bounds__(256) void k_cvt_bf16(const float* __restrict__ src,
                                                  unsigned short* __restrict__ dst) {
  long i4 = ((long)blockIdx.x * 256 + threadIdx.x) * 4;
  float4 v = *(const float4*)&src[i4];
  ushort4 o;
  o.x = f2bf(v.x); o.y = f2bf(v.y); o.z = f2bf(v.z); o.w = f2bf(v.w);
  *(ushort4*)&dst[i4] = o;
}

// ---------------------------------------------------------------------------
// GEMM: C[32768,1024] = A_bf16 @ W_bf16^T + bias (+residual), templated
// epilogue. Block tile 128x128, K-tile 32, 8 waves (4x2), wave tile 32x64.
// Double-buffered LDS: one s_barrier per K-step, global loads for tile k+1
// overlap the 8 WMMAs of tile k. Prefetch tile k+2 into L2.
// ---------------------------------------------------------------------------
template <int BF16OUT, int RESID>
__global__ __launch_bounds__(256) void k_gemm(
    const unsigned short* __restrict__ A, const unsigned short* __restrict__ W,
    const float* __restrict__ bias, const float* __restrict__ resid,
    unsigned short* __restrict__ out16, float* __restrict__ outf) {
  __shared__ __align__(16) unsigned short As[2][128 * 32];
  __shared__ __align__(16) unsigned short Bs[2][128 * 32];

  const int t    = threadIdx.x;
  const int lane = t & 31;
  const int w    = t >> 5;
  const int wm   = w & 3;        // wave row group (0..3)  -> 32 rows
  const int wn   = w >> 2;       // wave col group (0..1)  -> 64 cols
  const int bm   = blockIdx.y * 128;
  const int bn   = blockIdx.x * 128;
  const int l16  = lane & 15;
  const int half = lane >> 4;

  // per-thread global->LDS chunk mapping (2 x 16B per matrix per K-tile)
  const int r0 = t >> 2,          c0 = (t & 3) << 3;          // chunks 0..255
  const int r1 = (t + 256) >> 2,  c1 = ((t + 256) & 3) << 3;  // chunks 256..511
  const unsigned short* pA0 = A + (long)(bm + r0) * DMODEL + c0;
  const unsigned short* pA1 = A + (long)(bm + r1) * DMODEL + c1;
  const unsigned short* pB0 = W + (long)(bn + r0) * DMODEL + c0;
  const unsigned short* pB1 = W + (long)(bn + r1) * DMODEL + c1;

  v8f acc[2][4];
#pragma unroll
  for (int i = 0; i < 2; ++i)
#pragma unroll
    for (int j = 0; j < 4; ++j) acc[i][j] = (v8f)0.0f;

  // prologue: stage K-tile 0 into buffer 0
  {
    uint4 a0 = *(const uint4*)pA0, a1 = *(const uint4*)pA1;
    uint4 b0 = *(const uint4*)pB0, b1 = *(const uint4*)pB1;
    *(uint4*)&As[0][r0 * 32 + c0] = a0;
    *(uint4*)&As[0][r1 * 32 + c1] = a1;
    *(uint4*)&Bs[0][r0 * 32 + c0] = b0;
    *(uint4*)&Bs[0][r1 * 32 + c1] = b1;
  }

  int cur = 0;
  for (int kt = 0; kt < DMODEL; kt += 32) {
    __syncthreads();                       // buffer `cur` ready for all waves

    const bool has_next = (kt + 32) < DMODEL;
    uint4 a0, a1, b0, b1;
    if (has_next) {                        // overlap loads with WMMA below
      a0 = *(const uint4*)(pA0 + kt + 32);
      a1 = *(const uint4*)(pA1 + kt + 32);
      b0 = *(const uint4*)(pB0 + kt + 32);
      b1 = *(const uint4*)(pB1 + kt + 32);
    }
    if (kt + 64 < DMODEL) {                // keep tile k+2 warm in L2
      __builtin_prefetch(pA0 + kt + 64, 0, 1);
      __builtin_prefetch(pB0 + kt + 64, 0, 1);
    }

    v16bf af[2], bfr[4];
#pragma unroll
    for (int mt = 0; mt < 2; ++mt) {
      int m = wm * 32 + mt * 16 + l16;
      uint4 lo = *(const uint4*)&As[cur][m * 32 + half * 8];       // K 0-7 / 8-15
      uint4 hi = *(const uint4*)&As[cur][m * 32 + 16 + half * 8];  // K 16-23 / 24-31
      af[mt] = pack_frag(lo, hi);
    }
#pragma unroll
    for (int nt = 0; nt < 4; ++nt) {
      int nn = wn * 64 + nt * 16 + l16;
      uint4 lo = *(const uint4*)&Bs[cur][nn * 32 + half * 8];
      uint4 hi = *(const uint4*)&Bs[cur][nn * 32 + 16 + half * 8];
      bfr[nt] = pack_frag(lo, hi);
    }
#pragma unroll
    for (int mt = 0; mt < 2; ++mt)
#pragma unroll
      for (int nt = 0; nt < 4; ++nt)
        acc[mt][nt] = __builtin_amdgcn_wmma_f32_16x16x32_bf16(
            false, af[mt], false, bfr[nt], (short)0, acc[mt][nt], false, false);

    if (has_next) {                        // fill the other buffer
      int nxt = cur ^ 1;
      *(uint4*)&As[nxt][r0 * 32 + c0] = a0;
      *(uint4*)&As[nxt][r1 * 32 + c1] = a1;
      *(uint4*)&Bs[nxt][r0 * 32 + c0] = b0;
      *(uint4*)&Bs[nxt][r1 * 32 + c1] = b1;
      cur = nxt;
    }
  }

#pragma unroll
  for (int mt = 0; mt < 2; ++mt)
#pragma unroll
    for (int nt = 0; nt < 4; ++nt) {
      int gn = bn + wn * 64 + nt * 16 + l16;
      float bv = bias[gn];
#pragma unroll
      for (int r = 0; r < 8; ++r) {
        int gm = bm + wm * 32 + mt * 16 + r + half * 8;
        long o = (long)gm * DMODEL + gn;
        float v = acc[mt][nt][r] + bv;
        if (RESID) v += resid[o];
        if (BF16OUT) out16[o] = f2bf(v);
        else         outf[o]  = v;
      }
    }
}

// ---------------------------------------------------------------------------
// Per-row attention: one wave per row, 2 waves per block (68KB LDS).
//   scores[i,j] = sum_h q[h*64+i]*k[h*64+j] * 0.25   (K=16; K=16..31 region of
//                 Qt/Kt is zero-filled in LDS so padding is free ds_load_b128)
//   attn = softmax(scores) -> d_out (fp32) and LDS bf16
//   ctx[h*64+i] = sum_j attn[i,j]*v[h*64+j]          (K=64, two WMMA steps)
// ---------------------------------------------------------------------------
__global__ __launch_bounds__(64) void k_attn(
    const unsigned short* __restrict__ Q, const unsigned short* __restrict__ K,
    const unsigned short* __restrict__ V, float* __restrict__ attn,
    unsigned short* __restrict__ ctx) {
  __shared__ __align__(16) unsigned short Qt[2][64 * 32];   // Qt[i*32+h], h>=16 zero
  __shared__ __align__(16) unsigned short Kt[2][64 * 32];   // Kt[j*32+h], h>=16 zero
  __shared__ __align__(16) unsigned short Vs[2][1024];      // Vs[h*64+j]
  __shared__ __align__(16) float          Sf[2][64 * 64];
  __shared__ __align__(16) unsigned short Ab[2][64 * 64];   // bf16 attn

  const int w    = threadIdx.x >> 5;
  const int lane = threadIdx.x & 31;
  const int l16  = lane & 15;
  const int half = lane >> 4;
  const long n   = (long)blockIdx.x * 2 + w;

  const unsigned short* q = Q + n * DMODEL;
  const unsigned short* k = K + n * DMODEL;
  const unsigned short* v = V + n * DMODEL;

  const uint4 z = make_uint4(0, 0, 0, 0);

  // zero-fill the K=16..31 pad region once (2 rows per lane)
#pragma unroll
  for (int rr = 0; rr < 2; ++rr) {
    int row = lane * 2 + rr;
    *(uint4*)&Qt[w][row * 32 + 16] = z;
    *(uint4*)&Qt[w][row * 32 + 24] = z;
    *(uint4*)&Kt[w][row * 32 + 16] = z;
    *(uint4*)&Kt[w][row * 32 + 24] = z;
  }

  // stage Q,K transposed (i-major) and V row-copy into LDS
#pragma unroll
  for (int c = 0; c < 4; ++c) {
    int e0 = lane * 32 + c * 8;
    uint4 qv = *(const uint4*)&q[e0];
    uint4 kv = *(const uint4*)&k[e0];
    *(uint4*)&Vs[w][e0] = *(const uint4*)&v[e0];
    const unsigned short* qp = (const unsigned short*)&qv;
    const unsigned short* kp = (const unsigned short*)&kv;
#pragma unroll
    for (int tt = 0; tt < 8; ++tt) {
      int e = e0 + tt, h = e >> 6, i = e & 63;
      Qt[w][i * 32 + h] = qp[tt];
      Kt[w][i * 32 + h] = kp[tt];
    }
  }
  __syncthreads();

  const float scale = 0.25f;                 // 1/(TEMP*sqrt(H))

  // ---- scores = q k^T (K=16, zero-padded to 32 via LDS) ----
#pragma unroll
  for (int ti = 0; ti < 4; ++ti) {
    int i = ti * 16 + l16;
    uint4 alo = *(const uint4*)&Qt[w][i * 32 + half * 8];       // h 0-7 / 8-15
    uint4 ahi = *(const uint4*)&Qt[w][i * 32 + 16 + half * 8];  // zeros
    v16bf a = pack_frag(alo, ahi);
#pragma unroll
    for (int tj = 0; tj < 4; ++tj) {
      int j = tj * 16 + l16;
      uint4 b0 = *(const uint4*)&Kt[w][j * 32 + half * 8];       // h 0-7 / 8-15
      uint4 b1 = *(const uint4*)&Kt[w][j * 32 + 16 + half * 8];  // zeros
      v16bf b = pack_frag(b0, b1);
      v8f c = (v8f)0.0f;
      c = __builtin_amdgcn_wmma_f32_16x16x32_bf16(false, a, false, b,
                                                  (short)0, c, false, false);
#pragma unroll
      for (int r = 0; r < 8; ++r)
        Sf[w][(ti * 16 + r + half * 8) * 64 + tj * 16 + l16] = c[r] * scale;
    }
  }
  __syncthreads();

  // ---- softmax over j; write fp32 attn to d_out and bf16 copy to LDS ----
#pragma unroll
  for (int rr = 0; rr < 2; ++rr) {
    int i = lane + rr * 32;
    float mx = -3.4e38f;
    for (int j = 0; j < 64; ++j) mx = fmaxf(mx, Sf[w][i * 64 + j]);
    float s = 0.0f;
    for (int j = 0; j < 64; ++j) {
      float e = __expf(Sf[w][i * 64 + j] - mx);
      Sf[w][i * 64 + j] = e;
      s += e;
    }
    float inv = 1.0f / s;
    for (int j = 0; j < 64; ++j) {
      float p = Sf[w][i * 64 + j] * inv;
      attn[n * 4096 + i * 64 + j] = p;
      Ab[w][i * 64 + j] = f2bf(p);
    }
  }
  __syncthreads();

  // ---- ctx = attn @ v  (K=64 -> two WMMA K-steps) ----
#pragma unroll
  for (int ti = 0; ti < 4; ++ti) {
    v8f c = (v8f)0.0f;
#pragma unroll
    for (int k0 = 0; k0 < 64; k0 += 32) {
      int i = ti * 16 + l16;
      uint4 alo = *(const uint4*)&Ab[w][i * 64 + k0 + half * 8];
      uint4 ahi = *(const uint4*)&Ab[w][i * 64 + k0 + 16 + half * 8];
      v16bf a = pack_frag(alo, ahi);
      int h = l16;
      uint4 b0 = *(const uint4*)&Vs[w][h * 64 + k0 + half * 8];
      uint4 b1 = *(const uint4*)&Vs[w][h * 64 + k0 + 16 + half * 8];
      v16bf b = pack_frag(b0, b1);
      c = __builtin_amdgcn_wmma_f32_16x16x32_bf16(false, a, false, b,
                                                  (short)0, c, false, false);
    }
#pragma unroll
    for (int r = 0; r < 8; ++r) {
      int i = ti * 16 + r + half * 8;
      ctx[n * DMODEL + l16 * 64 + i] = f2bf(c[r]);   // ctx[n, h*64+i]
    }
  }
}

// ---------------------------------------------------------------------------
// LayerNorm over D=1024, one block (256 threads, 4 elems each) per row.
// ---------------------------------------------------------------------------
__global__ __launch_bounds__(256) void k_ln(const float* __restrict__ y,
                                            const float* __restrict__ gamma,
                                            const float* __restrict__ beta,
                                            float* __restrict__ out) {
  __shared__ float red[256];
  const int t = threadIdx.x;
  const long row = blockIdx.x;
  float4 v = *(const float4*)&y[row * DMODEL + t * 4];

  red[t] = v.x + v.y + v.z + v.w;
  __syncthreads();
  for (int o = 128; o > 0; o >>= 1) {
    if (t < o) red[t] += red[t + o];
    __syncthreads();
  }
  float mu = red[0] * (1.0f / DMODEL);
  __syncthreads();

  float dx = v.x - mu, dy = v.y - mu, dz = v.z - mu, dw = v.w - mu;
  red[t] = dx * dx + dy * dy + dz * dz + dw * dw;
  __syncthreads();
  for (int o = 128; o > 0; o >>= 1) {
    if (t < o) red[t] += red[t + o];
    __syncthreads();
  }
  float inv = rsqrtf(red[0] * (1.0f / DMODEL) + 1e-6f);

  float4 g = *(const float4*)&gamma[t * 4];
  float4 b = *(const float4*)&beta[t * 4];
  float4 o4;
  o4.x = dx * inv * g.x + b.x;
  o4.y = dy * inv * g.y + b.y;
  o4.z = dz * inv * g.z + b.z;
  o4.w = dw * inv * g.w + b.w;
  *(float4*)&out[row * DMODEL + t * 4] = o4;
}

// ---------------------------------------------------------------------------
extern "C" void kernel_launch(void* const* d_in, const int* in_sizes, int n_in,
                              void* d_out, int out_size, void* d_ws, size_t ws_size,
                              hipStream_t stream) {
  const long ND = (long)NROWS * DMODEL;   // 33,554,432
  const long DD = (long)DMODEL * DMODEL;  // 1,048,576

  const float* x     = (const float*)d_in[0];
  const float* Wq    = (const float*)d_in[1];
  const float* bq    = (const float*)d_in[2];
  const float* Wk    = (const float*)d_in[3];
  const float* bk    = (const float*)d_in[4];
  const float* Wv    = (const float*)d_in[5];
  const float* bv    = (const float*)d_in[6];
  const float* Wa    = (const float*)d_in[7];
  const float* ba    = (const float*)d_in[8];
  const float* gamma = (const float*)d_in[9];
  const float* beta  = (const float*)d_in[10];

  float* out  = (float*)d_out;       // [N, D]
  float* attn = out + ND;            // [N, 64, 64]

  char* p = (char*)d_ws;
  unsigned short* x16  = (unsigned short*)p; p += ND * 2;
  unsigned short* wq16 = (unsigned short*)p; p += DD * 2;
  unsigned short* wk16 = (unsigned short*)p; p += DD * 2;
  unsigned short* wv16 = (unsigned short*)p; p += DD * 2;
  unsigned short* wa16 = (unsigned short*)p; p += DD * 2;
  unsigned short* q16  = (unsigned short*)p; p += ND * 2;
  unsigned short* k16  = (unsigned short*)p; p += ND * 2;
  unsigned short* v16  = (unsigned short*)p; p += ND * 2;
  unsigned short* c16  = (unsigned short*)p; p += ND * 2;
  float*          ytmp = (float*)p;

  // 1) bf16 conversions (x + 4 weights)
  k_cvt_bf16<<<(int)(ND / 1024), 256, 0, stream>>>(x,  x16);
  k_cvt_bf16<<<(int)(DD / 1024), 256, 0, stream>>>(Wq, wq16);
  k_cvt_bf16<<<(int)(DD / 1024), 256, 0, stream>>>(Wk, wk16);
  k_cvt_bf16<<<(int)(DD / 1024), 256, 0, stream>>>(Wv, wv16);
  k_cvt_bf16<<<(int)(DD / 1024), 256, 0, stream>>>(Wa, wa16);

  // 2) Q,K,V projections (bf16 out, fp32 accumulate + bias)
  dim3 gg(DMODEL / 128, NROWS / 128);
  k_gemm<1, 0><<<gg, 256, 0, stream>>>(x16, wq16, bq, nullptr, q16, nullptr);
  k_gemm<1, 0><<<gg, 256, 0, stream>>>(x16, wk16, bk, nullptr, k16, nullptr);
  k_gemm<1, 0><<<gg, 256, 0, stream>>>(x16, wv16, bv, nullptr, v16, nullptr);

  // 3) per-row attention (scores WMMA -> softmax -> context WMMA)
  k_attn<<<NROWS / 2, 64, 0, stream>>>(q16, k16, v16, attn, c16);

  // 4) output projection + residual (fp32 out)
  k_gemm<0, 1><<<gg, 256, 0, stream>>>(c16, wa16, ba, x, nullptr, ytmp);

  // 5) LayerNorm
  k_ln<<<NROWS, 256, 0, stream>>>(ytmp, gamma, beta, out);
}
